// StreamSpatialGCN_26001732010354
// MI455X (gfx1250) — matrix operand order repaired and checked by reading, not compiled, not run on hardware
//
#include <hip/hip_runtime.h>

typedef __attribute__((ext_vector_type(16))) _Float16 v16h;
typedef __attribute__((ext_vector_type(8)))  float    v8f;

// ---------------- model dims ----------------
#define TDIM 300
#define VDIM 25
#define COUT 8
#define NSUB 3
#define TC_ATT 50      // attention K-chunk (t frames)
#define TC_FUS 10      // fusion chunk: 8*10 = 80 rows = 5 WMMA M-tiles
#define ACT (COUT*TDIM*VDIM)     // 60000
#define NTHREADS 256

struct BlockParamsGC {
  const float* A;                           // (3,25,25)
  const float* wa[3]; const float* ba[3];   // (2,cin),(2)
  const float* wb[3]; const float* bb[3];
  const float* wd[3]; const float* bd[3];   // (8,cin),(8)
  const float* fbn_g; const float* fbn_b;   // fusion BN
  const float* res_w; const float* res_b;   // l1 only
  const float* res_g; const float* res_bb;  // l1 only
  const float* tw; const float* tb;         // (8,8,9,1),(8)
  const float* tg; const float* tbb;        // tconv BN
};

struct ModelParamsGC {
  const float* x;
  const float* dbn_g; const float* dbn_b;   // (150)
  BlockParamsGC blk[3];
};

// LDS budget (bytes):
//  sX 120000 + sY 120000 + mask 1200 + P 2560 + colmax 128 + colsum 128
//  + BC16 6144 + scratch 20480 + weights 5200  = 275840  (< 320 KB/WGP)
#define SMEM_BYTES 275840

__global__ __launch_bounds__(NTHREADS, 1)
void agcn_fused_gfx1250(ModelParamsGC prm, float* __restrict__ out)
{
  const int tid  = threadIdx.x;
  const int wave = tid >> 5;
  const int lane = tid & 31;
  const int item = blockIdx.x;      // nm = n*2 + m
  const int n = item >> 1;
  const int m = item & 1;

  extern __shared__ char smem_raw[];
  char* cur = smem_raw;
  _Float16* sX   = (_Float16*)cur; cur += ACT*2;     // activation buffer (block input / block output)
  _Float16* sY   = (_Float16*)cur; cur += ACT*2;     // AGC output buffer
  float*    sMsk = (float*)cur;    cur += TDIM*4;
  float*    sP   = (float*)cur;    cur += 640*4;     // 25x25 attention logits
  float*    sCM  = (float*)cur;    cur += 32*4;      // per-column max
  float*    sCS  = (float*)cur;    cur += 32*4;      // per-column 1/sum
  _Float16* sBC  = (_Float16*)cur; cur += 3*1024*2;  // BC^T matrices, f16, [n=32][k=32] zero-padded
  float*    sScr = (float*)cur;    cur += 5120*4;    // phase-A a/b chunks; phase-B D chunks (f16 alias)
  _Float16* sD   = (_Float16*)sScr;                  // 3*80*32 halves = 15360 B <= 20480 B
  float* w      = (float*)cur;
  float* w_wa   = w;             // 48  [s][c][q<=8]
  float* w_ba   = w_wa  + 48;    // 8
  float* w_wb   = w_ba  + 8;     // 48
  float* w_bb   = w_wb  + 48;    // 8
  float* w_wd   = w_bb  + 8;     // 192 [s][o][q<=8]
  float* w_bds  = w_wd  + 192;   // 8   sum_i bd
  float* w_fg   = w_bds + 8;     // 8
  float* w_fb   = w_fg  + 8;     // 8
  float* w_rw   = w_fb  + 8;     // 64  res conv folded with res BN scale
  float* w_rb   = w_rw  + 64;    // 8
  float* w_tw   = w_rb  + 8;     // 576 [o][c][k]
  float* w_tb   = w_tw  + 576;   // 8
  float* w_tg   = w_tb  + 8;     // 8
  float* w_tbb  = w_tg  + 8;     // 8
  float* w_dbs  = w_tbb + 8;     // 150 data_bn scale
  float* w_dbb  = w_dbs + 150;   // 150 data_bn bias

  const float invs = rsqrtf(1.0f + 1e-5f);   // eval BN: g/sqrt(1+eps)
  const float* xg = prm.x;

  for (int i = tid; i < 150; i += NTHREADS) {
    w_dbs[i] = prm.dbn_g[i] * invs;
    w_dbb[i] = prm.dbn_b[i];
  }
  __syncthreads();

  // ---- frame mask: sum over (c,v) of raw x, fixed order (deterministic) ----
  for (int t = tid; t < TDIM; t += NTHREADS) {
    float s = 0.f;
    for (int c = 0; c < 3; c++)
      for (int v = 0; v < VDIM; v++)
        s += xg[((((size_t)n*3 + c)*TDIM + t)*VDIM + v)*2 + m];
    sMsk[t] = (s != 0.f) ? 1.f : 0.f;
  }
  // ---- load input, apply data_bn (channel = (m*V+v)*3+c) ----
  for (int idx = tid; idx < 3*TDIM*VDIM; idx += NTHREADS) {
    int c = idx / (TDIM*VDIM);
    int r = idx - c*(TDIM*VDIM);
    int t = r / VDIM;
    int v = r - t*VDIM;
    float xr = xg[((((size_t)n*3 + c)*TDIM + t)*VDIM + v)*2 + m];
    int ch = (m*VDIM + v)*3 + c;
    sX[idx] = (_Float16)(xr * w_dbs[ch] + w_dbb[ch]);
  }
  __syncthreads();

  int cin = 3;
  for (int blk = 0; blk < 3; blk++) {
    const BlockParamsGC bp = prm.blk[blk];

    // ---- stage this block's weights into LDS (pad cin to 8) ----
    for (int i = tid; i < 48; i += NTHREADS) {
      int s = i >> 4, c = (i >> 3) & 1, q = i & 7;
      w_wa[i] = (q < cin) ? bp.wa[s][c*cin + q] : 0.f;
      w_wb[i] = (q < cin) ? bp.wb[s][c*cin + q] : 0.f;
    }
    if (tid < 6) {
      w_ba[tid] = bp.ba[tid >> 1][tid & 1];
      w_bb[tid] = bp.bb[tid >> 1][tid & 1];
    }
    for (int i = tid; i < 192; i += NTHREADS) {
      int s = i >> 6, o = (i >> 3) & 7, q = i & 7;
      w_wd[i] = (q < cin) ? bp.wd[s][o*cin + q] : 0.f;
    }
    if (tid < 8) {
      w_bds[tid] = bp.bd[0][tid] + bp.bd[1][tid] + bp.bd[2][tid];
      w_fg[tid]  = bp.fbn_g[tid] * invs;
      w_fb[tid]  = bp.fbn_b[tid];
      w_tb[tid]  = bp.tb[tid];
      w_tg[tid]  = bp.tg[tid] * invs;
      w_tbb[tid] = bp.tbb[tid];
    }
    if (blk == 0) {
      if (tid < 8) {
        float g = bp.res_g[tid] * invs;
        w_rb[tid] = bp.res_b[tid] * g + bp.res_bb[tid];
      }
      for (int i = tid; i < 64; i += NTHREADS) {
        int o = i >> 3, q = i & 7;
        w_rw[i] = (q < cin) ? bp.res_w[o*cin + q] * (bp.res_g[o]*invs) : 0.f;
      }
    }
    for (int i = tid; i < 576; i += NTHREADS) w_tw[i] = bp.tw[i];
    __syncthreads();

    // ================= phase A: adaptive attention -> BC^T (f16, padded 32x32) =================
    int pv = tid / VDIM, pw = tid - pv*VDIM;   // owner of P[pv][pw] when tid<625
    for (int s = 0; s < NSUB; s++) {
      float preg = 0.f;
      for (int t0 = 0; t0 < TDIM; t0 += TC_ATT) {
        // a/b chunk: a[c][tl][v] at sScr[0..], b at sScr[2560..]
        for (int idx = tid; idx < 2*TC_ATT*VDIM; idx += NTHREADS) {
          int c  = idx / (TC_ATT*VDIM);
          int r  = idx - c*(TC_ATT*VDIM);
          int tl = r / VDIM;
          int v  = r - tl*VDIM;
          int t  = t0 + tl;
          float sa = w_ba[s*2 + c];
          float sb = w_bb[s*2 + c];
          for (int q = 0; q < cin; q++) {
            float xv = (float)sX[(q*TDIM + t)*VDIM + v];
            sa += w_wa[(s*2 + c)*8 + q] * xv;
            sb += w_wb[(s*2 + c)*8 + q] * xv;
          }
          sScr[idx]        = sa;
          sScr[2560 + idx] = sb;
        }
        __syncthreads();
        if (tid < VDIM*VDIM) {
          for (int tl = 0; tl < TC_ATT; tl++) {
            preg += sScr[tl*VDIM + pv]                 * sScr[2560 + tl*VDIM + pw];
            preg += sScr[TC_ATT*VDIM + tl*VDIM + pv]   * sScr[2560 + TC_ATT*VDIM + tl*VDIM + pw];
          }
        }
        __syncthreads();
      }
      if (tid < VDIM*VDIM) sP[pv*VDIM + pw] = preg * (1.f/600.f);  // /(ic*T)
      __syncthreads();
      if (tid < VDIM) {            // softmax over axis=-2 (rows v) per column w
        float mx = -3.4e38f;
        for (int v = 0; v < VDIM; v++) mx = fmaxf(mx, sP[v*VDIM + tid]);
        float ssum = 0.f;
        for (int v = 0; v < VDIM; v++) ssum += expf(sP[v*VDIM + tid] - mx);
        sCM[tid] = mx;
        sCS[tid] = 1.f / ssum;
      }
      for (int i = tid; i < 1024; i += NTHREADS) sBC[s*1024 + i] = (_Float16)0.f;
      __syncthreads();
      if (tid < VDIM*VDIM) {
        float ck = expf(sP[pv*VDIM + pw] - sCM[pw]) * sCS[pw];
        float bc = ck + bp.A[s*VDIM*VDIM + pv*VDIM + pw] + 1e-6f;   // + A_adaptive + anti_zero
        // store TRANSPOSED: [n=w][k=v] so each WMMA lane reads 16 contiguous halves
        sBC[s*1024 + pw*32 + pv] = (_Float16)bc;
      }
      __syncthreads();
    }

    // ================= phase B: fusion = sum_i (Wd_i X) @ BC_i  via WMMA =================
    for (int t0 = 0; t0 < TDIM; t0 += TC_FUS) {
      // D[s][mrow=o*10+tl][k=v], f16, K padded to 32
      for (int idx = tid; idx < NSUB*80*32; idx += NTHREADS) {
        int s  = idx / (80*32);
        int r  = idx - s*(80*32);
        int mm = r >> 5;
        int k  = r & 31;
        float val = 0.f;
        if (k < VDIM) {
          int o  = mm / TC_FUS;
          int tl = mm - o*TC_FUS;
          int t  = t0 + tl;
          for (int c = 0; c < cin; c++)
            val += w_wd[(s*8 + o)*8 + c] * (float)sX[(c*TDIM + t)*VDIM + k];
        }
        sD[idx] = (_Float16)val;
      }
      __syncthreads();
      // 5 M-tiles x 2 N-tiles = 10 wave tasks
      for (int task = wave; task < 10; task += 8) {
        int mt = task >> 1;
        int nt = task & 1;
        v8f acc = {0.f,0.f,0.f,0.f,0.f,0.f,0.f,0.f};
        int row = mt*16 + (lane & 15);
        int col = nt*16 + (lane & 15);
        int kha = (lane >> 4) << 3;    // unused: A is loaded with khb, hi lanes hold K+16 like B
        int khb = (lane >> 4) << 4;    // B: hi lanes hold K+16 (dense 32x16)
        for (int s = 0; s < NSUB; s++) {
          v16h af, bf;
#pragma unroll
          for (int e = 0; e < 16; e++) {
            int j = e >> 1;
            int K = e + khb; (void)j;
            af[e] = sD[(s*80 + row)*32 + K];
          }
          // BC stored transposed [n][k]: lane reads K-contiguous run -> 2x ds_load_b128
#pragma unroll
          for (int e = 0; e < 16; e++)
            bf[e] = sBC[s*1024 + col*32 + khb + e];
          acc = __builtin_amdgcn_wmma_f32_16x16x32_f16(false, af, false, bf,
                                                       (short)0, acc, false, false);
        }
        // D layout: VGPR r -> M=r (lanes 0-15) / M=r+8 (lanes 16-31), N=lane&15
#pragma unroll
        for (int r8 = 0; r8 < 8; r8++) {
          int mm = mt*16 + r8 + ((lane >> 4) << 3);
          if (col < VDIM) {
            int o  = mm / TC_FUS;
            int tl = mm - o*TC_FUS;
            int t  = t0 + tl;
            float val = acc[r8] + w_bds[o];          // conv_d biases (summed over subsets)
            val = val * w_fg[o] + w_fb[o];           // fusion BN
            float res;
            if (blk == 0) {                          // 1x1 res conv + BN (folded)
              res = w_rb[o];
              for (int c = 0; c < cin; c++)
                res += w_rw[o*8 + c] * (float)sX[(c*TDIM + t)*VDIM + col];
            } else {                                 // identity residual inside AGC
              res = (float)sX[(o*TDIM + t)*VDIM + col];
            }
            val += res;
            sY[(o*TDIM + t)*VDIM + col] = (_Float16)fmaxf(val, 0.f);
          }
        }
      }
      __syncthreads();
    }

    // ================= phase C: 9x1 temporal conv + BN (+identity) + ReLU -> sX =================
    for (int idx = tid; idx < ACT; idx += NTHREADS) {
      int o = idx / (TDIM*VDIM);
      int r = idx - o*(TDIM*VDIM);
      int t = r / VDIM;
      int v = r - t*VDIM;
      float acc = w_tb[o];
      for (int c = 0; c < COUT; c++) {
#pragma unroll
        for (int k = 0; k < 9; k++) {
          int tt = t + k - 4;
          if (tt >= 0 && tt < TDIM)
            acc += w_tw[(o*8 + c)*9 + k] * (float)sY[(c*TDIM + tt)*VDIM + v];
        }
      }
      acc = acc * w_tg[o] + w_tbb[o];
      if (blk > 0) acc += (float)sX[idx];   // block-level identity residual (l2,l3)
      sX[idx] = (_Float16)fmaxf(acc, 0.f);
    }
    __syncthreads();
    cin = COUT;
  }

  // ---- masked output: (N, 8, T, V, M) ----
  for (int idx = tid; idx < ACT; idx += NTHREADS) {
    int o = idx / (TDIM*VDIM);
    int r = idx - o*(TDIM*VDIM);
    int t = r / VDIM;
    int v = r - t*VDIM;
    out[(((((size_t)n*COUT + o)*TDIM + t)*VDIM + v) << 1) + m] =
        (float)sX[idx] * sMsk[t];
  }
}

extern "C" void kernel_launch(void* const* d_in, const int* in_sizes, int n_in,
                              void* d_out, int out_size, void* d_ws, size_t ws_size,
                              hipStream_t stream) {
  (void)d_ws; (void)ws_size; (void)out_size;
  const float* P[128];
  int cnt = (n_in < 128) ? n_in : 128;
  for (int i = 0; i < cnt; i++) P[i] = (const float*)d_in[i];

  // locate x (largest array); params may come before or after it
  int xi = 0;
  for (int i = 1; i < cnt; i++) if (in_sizes[i] > in_sizes[xi]) xi = i;
  int base = (xi == 0) ? 1 : 0;

  // schema probe: params order = [data_bn(2), l1.A, <next>...]
  //   insertion order: next leaf = conv_a[0].w (size 6)
  //   jax-sorted pytree: next leaf = agc.bn.b (size 8)
  bool sorted_order = (base + 3 < cnt) && (in_sizes[base + 3] == 8);

  ModelParamsGC prm{};
  prm.x = P[xi];
  int idx = base;
  if (!sorted_order) { prm.dbn_g = P[idx++]; prm.dbn_b = P[idx++]; }
  else               { prm.dbn_b = P[idx++]; prm.dbn_g = P[idx++]; }

  for (int b = 0; b < 3; b++) {
    BlockParamsGC& B = prm.blk[b];
    bool has_res = (b == 0);
    if (!sorted_order) {
      // agc: A, conv_a{w,b}x3, conv_b{w,b}x3, conv_d{w,b}x3, bn{g,b}, [res: conv{w,b}, bn{g,b}]
      B.A = P[idx++];
      for (int i = 0; i < 3; i++) { B.wa[i] = P[idx++]; B.ba[i] = P[idx++]; }
      for (int i = 0; i < 3; i++) { B.wb[i] = P[idx++]; B.bb[i] = P[idx++]; }
      for (int i = 0; i < 3; i++) { B.wd[i] = P[idx++]; B.bd[i] = P[idx++]; }
      B.fbn_g = P[idx++]; B.fbn_b = P[idx++];
      if (has_res) { B.res_w = P[idx++]; B.res_b = P[idx++]; B.res_g = P[idx++]; B.res_bb = P[idx++]; }
      // block: tconv_w, tconv_b, tconv_bn{g,b}
      B.tw = P[idx++]; B.tb = P[idx++]; B.tg = P[idx++]; B.tbb = P[idx++];
    } else {
      // agc sorted: A, bn{b,g}, conv_a{b,w}x3, conv_b, conv_d, [res: bn{b,g}, conv{b,w}]
      B.A = P[idx++];
      B.fbn_b = P[idx++]; B.fbn_g = P[idx++];
      for (int i = 0; i < 3; i++) { B.ba[i] = P[idx++]; B.wa[i] = P[idx++]; }
      for (int i = 0; i < 3; i++) { B.bb[i] = P[idx++]; B.wb[i] = P[idx++]; }
      for (int i = 0; i < 3; i++) { B.bd[i] = P[idx++]; B.wd[i] = P[idx++]; }
      if (has_res) { B.res_bb = P[idx++]; B.res_g = P[idx++]; B.res_b = P[idx++]; B.res_w = P[idx++]; }
      // block sorted: tconv_b, tconv_bn{b,g}, tconv_w
      B.tb = P[idx++]; B.tbb = P[idx++]; B.tg = P[idx++]; B.tw = P[idx++];
    }
  }

  int items = in_sizes[xi] / (3 * TDIM * VDIM * 2);   // N*M = 512
  (void)hipFuncSetAttribute((const void*)agcn_fused_gfx1250,
                            hipFuncAttributeMaxDynamicSharedMemorySize,
                            (int)SMEM_BYTES);
  agcn_fused_gfx1250<<<items, NTHREADS, SMEM_BYTES, stream>>>(prm, (float*)d_out);
}
